// VAE_60361470378654
// MI455X (gfx1250) — matrix operand order repaired and checked by reading, not compiled, not run on hardware
//
#include <hip/hip_runtime.h>
#include <hip/hip_bf16.h>
#include <math.h>

typedef __attribute__((ext_vector_type(16))) _Float16 v16h;
typedef __attribute__((ext_vector_type(8)))  _Float16 v8h;
typedef __attribute__((ext_vector_type(8)))  float    v8f;

#define LSDIM 30
#define BATCH 1024
#define HW    784            // 28*28 interior pixels
#define NPIX  (BATCH * HW)   // 802816
#define PW    30             // padded width/height
#define PHW   (PW * PW)      // 900 padded pixels per image

__device__ __forceinline__ float lrelu(float v) { return v > 0.f ? v : 0.01f * v; }

// flat interior pixel -> element offset in padded [B][30][30][C] tensor
__device__ __forceinline__ size_t pad_off(int p, int C) {
    int b = p / HW;
    int rem = p % HW;
    int y = rem / 28, x = rem % 28;
    return ((size_t)(b * PW + (y + 1)) * PW + (x + 1)) * C;
}

// ---------------------------------------------------------------------------
// Weight fragment packer: conv_w f32 [O][Cin][3][3] -> f16 A-fragment buffer
// frag[((t*G + g)*4 + m)*512 + lane*16 + e]  (G = Cin/32)
// A-matrix 16x32 f16 layout (contiguous K, matches B fragment):
//   lane<16 : M=lane,    e0..15 -> K 0..15
//   lane>=16: M=lane-16, e0..15 -> K 16..31
// ---------------------------------------------------------------------------
__global__ void vae_pack_wfrag(const float* __restrict__ w, _Float16* __restrict__ frag,
                               int cin, int total) {
    int idx = blockIdx.x * blockDim.x + threadIdx.x;
    if (idx >= total) return;
    int G    = cin >> 5;
    int e    = idx & 15;
    int lane = (idx >> 4) & 31;
    int m    = (idx >> 9) & 3;
    int g    = (idx >> 11) % G;
    int t    = (idx >> 11) / G;
    int M    = lane & 15;
    int K    = (lane & 16) + e;
    int O  = m * 16 + M;
    int I  = g * 32 + K;
    int ky = t / 3, kx = t % 3;
    frag[idx] = (_Float16)w[((O * cin + I) * 3 + ky) * 3 + kx];
}

// Zero the 1-pixel halo border of a padded [B][30][30][C] f16 tensor.
__global__ void vae_zero_border(_Float16* __restrict__ buf, int C, int total) {
    int idx = blockIdx.x * blockDim.x + threadIdx.x;
    if (idx >= total) return;                    // total = BATCH * 116 * C
    int c = idx % C;
    int r = idx / C;
    int b = r / 116;
    int i = r % 116;
    int y, x;
    if (i < 30)      { y = 0;          x = i; }
    else if (i < 60) { y = 29;         x = i - 30; }
    else if (i < 88) { y = i - 60 + 1; x = 0; }
    else             { y = i - 88 + 1; x = 29; }
    buf[((size_t)(b * PW + y) * PW + x) * C + c] = (_Float16)0.f;
}

// ---------------------------------------------------------------------------
// Decoder conv 3x3 pad=1, zero-padded NHWC f16.
// Register-blocked 4x4: one wave = 64 oc x (same 16-pixel tile in 4
// consecutive images).  The 4 N-tiles sit at constant offsets i*PHW*C from a
// single base pointer, so every load in the loop is one address register +
// an immediate offset (no in-loop address VALU, <256 VGPRs).
// ---------------------------------------------------------------------------
template <int CIN>
__global__ __launch_bounds__(256) void vae_conv3x3_wmma(
    const _Float16* __restrict__ in,     // [B*900][CIN], zero halo
    const _Float16* __restrict__ wfrag,  // packed A fragments
    const float*    __restrict__ bias,   // [64]
    _Float16*       __restrict__ out)    // [B*900][64], interior written
{
    constexpr int G = CIN / 32;
    constexpr int BSTRIDE_IN  = PHW * CIN;   // elements between images (input)
    constexpr int BSTRIDE_OUT = PHW * 64;    // elements between images (output)

    const int lane = threadIdx.x & 31;
    const int wave = threadIdx.x >> 5;
    const int q    = blockIdx.x * 8 + wave;    // wave id in [0, 12544)
    const int b0   = (q / 49) * 4;             // first of 4 consecutive images
    const int tt   = q % 49;                   // 16-pixel tile within image
    const int col  = lane & 15;
    const int koff = lane & 16;                // B-frag K offset: 0 or 16

    const int p0 = b0 * HW + tt * 16 + col;    // pixel in image b0
    const _Float16* src = in + pad_off(p0, CIN) + koff;

    v8f a00 = {}, a01 = {}, a02 = {}, a03 = {};
    v8f a10 = {}, a11 = {}, a12 = {}, a13 = {};
    v8f a20 = {}, a21 = {}, a22 = {}, a23 = {};
    v8f a30 = {}, a31 = {}, a32 = {}, a33 = {};

    const _Float16* wl = wfrag + lane * 16;

#pragma unroll
    for (int t = 0; t < 9; ++t) {
        const int toff = ((t / 3 - 1) * PW + (t % 3 - 1)) * CIN;  // constant
#pragma unroll
        for (int g = 0; g < G; ++g) {
            const int o = toff + g * 32;                          // constant
            v16h b0f = *(const v16h*)(src + o + 0 * BSTRIDE_IN);
            v16h b1f = *(const v16h*)(src + o + 1 * BSTRIDE_IN);
            v16h b2f = *(const v16h*)(src + o + 2 * BSTRIDE_IN);
            v16h b3f = *(const v16h*)(src + o + 3 * BSTRIDE_IN);
            const _Float16* wf = wl + (size_t)((t * G + g) * 4) * 512;
            v16h w0 = *(const v16h*)(wf);
            v16h w1 = *(const v16h*)(wf + 512);
            v16h w2 = *(const v16h*)(wf + 1024);
            v16h w3 = *(const v16h*)(wf + 1536);
#define VW(ACC, WA, BB) \
            ACC = __builtin_amdgcn_wmma_f32_16x16x32_f16(false, WA, false, BB, (short)0, ACC, false, false);
            VW(a00, w0, b0f) VW(a10, w1, b0f) VW(a20, w2, b0f) VW(a30, w3, b0f)
            VW(a01, w0, b1f) VW(a11, w1, b1f) VW(a21, w2, b1f) VW(a31, w3, b1f)
            VW(a02, w0, b2f) VW(a12, w1, b2f) VW(a22, w2, b2f) VW(a32, w3, b2f)
            VW(a03, w0, b3f) VW(a13, w1, b3f) VW(a23, w2, b3f) VW(a33, w3, b3f)
#undef VW
        }
    }

    // C/D layout: VGPR r -> M = r + (lane>=16 ? 8 : 0), N = lane%16
    const int rowoff = (lane & 16) ? 8 : 0;
    const v8f bv0 = *(const v8f*)(bias + 0  + rowoff);
    const v8f bv1 = *(const v8f*)(bias + 16 + rowoff);
    const v8f bv2 = *(const v8f*)(bias + 32 + rowoff);
    const v8f bv3 = *(const v8f*)(bias + 48 + rowoff);

    _Float16* ob = out + pad_off(p0, 64) + rowoff;

#define VAE_EPI(ACC, BV, N, M)                                  \
    {                                                           \
        v8h ov;                                                 \
        _Pragma("unroll") for (int r = 0; r < 8; ++r) {         \
            float v = ACC[r] + BV[r];                           \
            ov[r] = (_Float16)(v > 0.f ? v : 0.01f * v);        \
        }                                                       \
        *(v8h*)(ob + (N)*BSTRIDE_OUT + (M)*16) = ov;            \
    }
    VAE_EPI(a00, bv0, 0, 0) VAE_EPI(a10, bv1, 0, 1) VAE_EPI(a20, bv2, 0, 2) VAE_EPI(a30, bv3, 0, 3)
    VAE_EPI(a01, bv0, 1, 0) VAE_EPI(a11, bv1, 1, 1) VAE_EPI(a21, bv2, 1, 2) VAE_EPI(a31, bv3, 1, 3)
    VAE_EPI(a02, bv0, 2, 0) VAE_EPI(a12, bv1, 2, 1) VAE_EPI(a22, bv2, 2, 2) VAE_EPI(a32, bv3, 2, 3)
    VAE_EPI(a03, bv0, 3, 0) VAE_EPI(a13, bv1, 3, 1) VAE_EPI(a23, bv2, 3, 2) VAE_EPI(a33, bv3, 3, 3)
#undef VAE_EPI
}

// ---------------------------------------------------------------------------
// Encoder (tiny — direct VALU kernels, NCHW f32)
// ---------------------------------------------------------------------------
__global__ void vae_enc1(const float* __restrict__ x, const float* __restrict__ w,
                         const float* __restrict__ bias, float* __restrict__ out, int total) {
    int idx = blockIdx.x * blockDim.x + threadIdx.x;
    if (idx >= total) return;                 // [B,8,13,13]
    int b  = idx / (8 * 169);
    int r  = idx % (8 * 169);
    int oc = r / 169;
    int oy = (r % 169) / 13, ox = r % 13;
    const float* xb = x + (size_t)b * HW;
    float m = -1e30f;
    for (int py = 0; py < 2; ++py)
        for (int px = 0; px < 2; ++px) {
            int iy0 = oy * 2 + py, ix0 = ox * 2 + px;   // 26x26 conv grid
            float acc = bias[oc];
            for (int ky = 0; ky < 3; ++ky)
                for (int kx = 0; kx < 3; ++kx)
                    acc += xb[(iy0 + ky) * 28 + ix0 + kx] * w[(oc * 9) + ky * 3 + kx];
            acc = lrelu(acc);
            m = fmaxf(m, acc);
        }
    out[idx] = m;
}

__global__ void vae_enc2(const float* __restrict__ in, const float* __restrict__ w,
                         const float* __restrict__ bias, float* __restrict__ out, int total) {
    int idx = blockIdx.x * blockDim.x + threadIdx.x;
    if (idx >= total) return;                 // in [B,8,13,13] -> out [B,16,6,6]
    int b  = idx / (16 * 36);
    int r  = idx % (16 * 36);
    int oc = r / 36;
    int oy = (r % 36) / 6, ox = r % 6;
    float m = -1e30f;
    for (int py = 0; py < 2; ++py)
        for (int px = 0; px < 2; ++px) {
            int iy0 = oy * 2 + py, ix0 = ox * 2 + px;   // 12x12 conv grid
            float acc = bias[oc];
            for (int ic = 0; ic < 8; ++ic)
                for (int ky = 0; ky < 2; ++ky)
                    for (int kx = 0; kx < 2; ++kx)
                        acc += in[((b * 8 + ic) * 13 + iy0 + ky) * 13 + ix0 + kx] *
                               w[((oc * 8 + ic) * 2 + ky) * 2 + kx];
            acc = lrelu(acc);
            m = fmaxf(m, acc);
        }
    out[idx] = m;
}

__global__ void vae_enc3(const float* __restrict__ in, const float* __restrict__ w,
                         const float* __restrict__ bias, float* __restrict__ out, int total) {
    int idx = blockIdx.x * blockDim.x + threadIdx.x;
    if (idx >= total) return;                 // in [B,16,6,6] -> out [B,32,4,4]
    int b  = idx / (32 * 16);
    int r  = idx % (32 * 16);
    int oc = r / 16;
    int oy = (r % 16) / 4, ox = r % 4;
    float acc = bias[oc];
    for (int ic = 0; ic < 16; ++ic)
        for (int ky = 0; ky < 3; ++ky)
            for (int kx = 0; kx < 3; ++kx)
                acc += in[((b * 16 + ic) * 6 + oy + ky) * 6 + ox + kx] *
                       w[((oc * 16 + ic) * 3 + ky) * 3 + kx];
    out[idx] = lrelu(acc);
}

__global__ void vae_enc4(const float* __restrict__ in, const float* __restrict__ w,
                         const float* __restrict__ bias, float* __restrict__ flat, int total) {
    int idx = blockIdx.x * blockDim.x + threadIdx.x;
    if (idx >= total) return;                 // in [B,32,4,4] -> flat [B,256] (c*4+y*2+x)
    int b  = idx / 256;
    int r  = idx % 256;
    int oc = r >> 2;
    int oy = (r >> 1) & 1, ox = r & 1;
    float acc = bias[oc];
    for (int ic = 0; ic < 32; ++ic)
        for (int ky = 0; ky < 3; ++ky)
            for (int kx = 0; kx < 3; ++kx)
                acc += in[((b * 32 + ic) * 4 + oy + ky) * 4 + ox + kx] *
                       w[((oc * 32 + ic) * 3 + ky) * 3 + kx];
    flat[idx] = lrelu(acc);
}

// FC (mu, logvar) + reparameterize -> writes mu/logvar/z
__global__ void vae_fc_reparam(const float* __restrict__ flat,
                               const float* __restrict__ mw, const float* __restrict__ mb,
                               const float* __restrict__ vw, const float* __restrict__ vb,
                               const float* __restrict__ eps,
                               float* __restrict__ mu_o, float* __restrict__ lv_o,
                               float* __restrict__ z_o, int total) {
    int idx = blockIdx.x * blockDim.x + threadIdx.x;
    if (idx >= total) return;                 // B*LSDIM
    int b = idx / LSDIM, j = idx % LSDIM;
    const float* f = flat + (size_t)b * 256;
    float mu = mb[j], lv = vb[j];
    for (int k = 0; k < 256; ++k) {
        float fv = f[k];
        mu += fv * mw[j * 256 + k];
        lv += fv * vw[j * 256 + k];
    }
    float z = eps[idx] * expf(0.5f * lv) + mu;
    mu_o[idx] = mu;
    lv_o[idx] = lv;
    z_o[idx]  = z;
}

// Decoder input into padded interior: [B*900][32] f16
// (c0 = step[h], c1 = step[w], c2.. = z)
__global__ void vae_build_d0(const float* __restrict__ z, _Float16* __restrict__ d0, int total) {
    int idx = blockIdx.x * blockDim.x + threadIdx.x;
    if (idx >= total) return;                 // NPIX * 32
    int c = idx & 31;
    int p = idx >> 5;
    int b = p / HW;
    int rem = p % HW;
    int h = rem / 28, w = rem % 28;
    float v;
    if (c == 0)      v = -1.f + (2.f / 27.f) * (float)h;
    else if (c == 1) v = -1.f + (2.f / 27.f) * (float)w;
    else             v = z[b * LSDIM + (c - 2)];
    d0[((size_t)(b * PW + (h + 1)) * PW + (w + 1)) * 32 + c] = (_Float16)v;
}

// conv9: 1x1, 64 -> 1, + lrelu, f32 output (reads padded interior)
__global__ void vae_conv9(const _Float16* __restrict__ act, const float* __restrict__ w,
                          const float* __restrict__ bias, float* __restrict__ out, int total) {
    int p = blockIdx.x * blockDim.x + threadIdx.x;
    if (p >= total) return;
    const _Float16* a = act + pad_off(p, 64);
    float acc = bias[0];
#pragma unroll
    for (int c = 0; c < 64; ++c) acc += (float)a[c] * w[c];
    out[p] = lrelu(acc);
}

// ---------------------------------------------------------------------------
extern "C" void kernel_launch(void* const* d_in, const int* in_sizes, int n_in,
                              void* d_out, int out_size, void* d_ws, size_t ws_size,
                              hipStream_t stream) {
    (void)in_sizes; (void)n_in; (void)out_size; (void)ws_size;

    const float* x   = (const float*)d_in[0];
    const float* eps = (const float*)d_in[1];
    const float* c1w = (const float*)d_in[2];
    const float* c1b = (const float*)d_in[3];
    const float* c2w = (const float*)d_in[4];
    const float* c2b = (const float*)d_in[5];
    const float* c3w = (const float*)d_in[6];
    const float* c3b = (const float*)d_in[7];
    const float* c4w = (const float*)d_in[8];
    const float* c4b = (const float*)d_in[9];
    const float* mw  = (const float*)d_in[10];
    const float* mb  = (const float*)d_in[11];
    const float* vw  = (const float*)d_in[12];
    const float* vb  = (const float*)d_in[13];
    const float* c5w = (const float*)d_in[14];
    const float* c5b = (const float*)d_in[15];
    const float* c6w = (const float*)d_in[16];
    const float* c6b = (const float*)d_in[17];
    const float* c7w = (const float*)d_in[18];
    const float* c7b = (const float*)d_in[19];
    const float* c8w = (const float*)d_in[20];
    const float* c8b = (const float*)d_in[21];
    const float* c9w = (const float*)d_in[22];
    const float* c9b = (const float*)d_in[23];

    float* out_d  = (float*)d_out;                 // [NPIX]
    float* out_mu = out_d + NPIX;                  // [B*LSDIM]
    float* out_lv = out_mu + BATCH * LSDIM;
    float* out_z  = out_lv + BATCH * LSDIM;

    // --- workspace carving ---
    char* wsp = (char*)d_ws;
    auto carve = [&](size_t bytes) {
        char* p = wsp;
        wsp += (bytes + 255) & ~(size_t)255;
        return p;
    };
    float*    e1   = (float*)   carve((size_t)BATCH * 8 * 169 * 4);
    float*    e2   = (float*)   carve((size_t)BATCH * 16 * 36 * 4);
    float*    e3   = (float*)   carve((size_t)BATCH * 32 * 16 * 4);
    float*    flat = (float*)   carve((size_t)BATCH * 256 * 4);
    _Float16* f5   = (_Float16*)carve((size_t)9 * 1 * 4 * 512 * 2);
    _Float16* f6   = (_Float16*)carve((size_t)9 * 2 * 4 * 512 * 2);
    _Float16* f7   = (_Float16*)carve((size_t)9 * 2 * 4 * 512 * 2);
    _Float16* f8   = (_Float16*)carve((size_t)9 * 2 * 4 * 512 * 2);
    _Float16* d0   = (_Float16*)carve((size_t)BATCH * PHW * 32 * 2);   // padded
    _Float16* actA = (_Float16*)carve((size_t)BATCH * PHW * 64 * 2);   // padded
    _Float16* actB = (_Float16*)carve((size_t)BATCH * PHW * 64 * 2);   // padded

    const int T = 256;
    auto nb = [](int n, int t) { return (n + t - 1) / t; };

    // halo zeroing (must run every call; ws contents are not preserved)
    int zb32 = BATCH * 116 * 32;
    int zb64 = BATCH * 116 * 64;
    vae_zero_border<<<nb(zb32, T), T, 0, stream>>>(d0,   32, zb32);
    vae_zero_border<<<nb(zb64, T), T, 0, stream>>>(actA, 64, zb64);
    vae_zero_border<<<nb(zb64, T), T, 0, stream>>>(actB, 64, zb64);

    // weight fragment packing (independent of data path)
    vae_pack_wfrag<<<nb(18432, T), T, 0, stream>>>(c5w, f5, 32, 18432);
    vae_pack_wfrag<<<nb(36864, T), T, 0, stream>>>(c6w, f6, 64, 36864);
    vae_pack_wfrag<<<nb(36864, T), T, 0, stream>>>(c7w, f7, 64, 36864);
    vae_pack_wfrag<<<nb(36864, T), T, 0, stream>>>(c8w, f8, 64, 36864);

    // encoder
    int n1 = BATCH * 8 * 169;
    vae_enc1<<<nb(n1, T), T, 0, stream>>>(x, c1w, c1b, e1, n1);
    int n2 = BATCH * 16 * 36;
    vae_enc2<<<nb(n2, T), T, 0, stream>>>(e1, c2w, c2b, e2, n2);
    int n3 = BATCH * 32 * 16;
    vae_enc3<<<nb(n3, T), T, 0, stream>>>(e2, c3w, c3b, e3, n3);
    int n4 = BATCH * 256;
    vae_enc4<<<nb(n4, T), T, 0, stream>>>(e3, c4w, c4b, flat, n4);

    // fc + reparameterize (writes mu/logvar/z outputs)
    int nf = BATCH * LSDIM;
    vae_fc_reparam<<<nb(nf, T), T, 0, stream>>>(flat, mw, mb, vw, vb, eps,
                                                out_mu, out_lv, out_z, nf);

    // decoder input (coordinate planes + broadcast z): L+2 == 32 channels
    int nd0 = NPIX * 32;
    vae_build_d0<<<nb(nd0, T), T, 0, stream>>>(out_z, d0, nd0);

    // decoder: 4x WMMA implicit-GEMM conv layers
    // 12544 waves (256 image-quads x 49 tiles), 8 waves/block -> 1568 blocks
    const int CONV_BLOCKS = NPIX / 64 / 8;   // 1568
    vae_conv3x3_wmma<32><<<CONV_BLOCKS, 256, 0, stream>>>(d0,   f5, c5b, actA);
    vae_conv3x3_wmma<64><<<CONV_BLOCKS, 256, 0, stream>>>(actA, f6, c6b, actB);
    vae_conv3x3_wmma<64><<<CONV_BLOCKS, 256, 0, stream>>>(actB, f7, c7b, actA);
    vae_conv3x3_wmma<64><<<CONV_BLOCKS, 256, 0, stream>>>(actA, f8, c8b, actB);

    // 1x1 reduction to image output
    vae_conv9<<<nb(NPIX, T), T, 0, stream>>>(actB, c9w, c9b, out_d, NPIX);
}
